// EHFF_85581518340272
// MI455X (gfx1250) — compile-verified, matrix operand order checked by reading, not run on hardware
//
#include <hip/hip_runtime.h>
#include <hip/hip_bf16.h>

// ---------------------------------------------------------------------------
// EHFF pipeline for gfx1250 (MI455X): fc1 GEMM -> 2x (offset conv, bilinear
// deform gather, deform GEMM, conv) -> LN+GELU -> fc2 GEMM.
// Heavy matmuls via V_WMMA_F32_16X16X32_F16 (f16 in, f32 accum), NHWC layout.
// Block = 256 threads (8 wave32s); tile = 64 oc x 128 px; each wave owns a
// 32x32 quad (2x2 WMMA subtiles); LDS ping-pong, 1 barrier / K-chunk.
// ---------------------------------------------------------------------------

typedef __attribute__((ext_vector_type(16))) _Float16 v16h;
typedef __attribute__((ext_vector_type(8)))  _Float16 half8;
typedef __attribute__((ext_vector_type(8)))  float    v8f;

#define LSTR 40   // LDS tile row stride in halfs (80B: 16B-aligned, bank-spread)

// Per-lane WMMA fragment load from an LDS tile stored row-major (row, k) with
// stride LSTR. CDNA5 16-bit A(m,k) / B-as-(n,k) layouts are symmetric:
//   lane = row%16 + 16*((k>>3)&1), vgpr = ((k&7)>>1)+4*(k>>4), half = k&1
// => lane l reads halfs [fs..fs+7] and [16+fs..16+fs+7] of row (l&15), fs=(l>>4)*8.
__device__ __forceinline__ v16h frag_ld(const _Float16* tile, int rowOff, int lane) {
  const int fr = lane & 15;
  const int fs = (lane >> 4) << 3;
  const _Float16* p = tile + (rowOff + fr) * LSTR + fs;
  union { v16h v; uint4 q[2]; } u;
  u.q[0] = *(const uint4*)(p);
  u.q[1] = *(const uint4*)(p + 16);
  return u.v;
}

__device__ __forceinline__ v8f wmma_f16(v16h a, v16h b, v8f c) {
  return __builtin_amdgcn_wmma_f32_16x16x32_f16(false, a, false, b,
                                                (short)0, c, false, false);
}

// ---------------------------------------------------------------------------
// Unified implicit-GEMM conv / GEMM kernel (templated on kernel geometry).
//   out[t, oc] = bias[oc] + sum_{kpos,c} w[oc][kpos*Cin+c] * in[neigh(t,kpos)*inStride + c]
// KK==1/KS==1/PAD==0 degenerates to a plain GEMM (fc1/fc2/deform-matmul).
// Loads are unconditional from clamped addresses; OOB masked by VALU selects,
// so the K-loop is straight-line and barriers are trivially uniform.
// ---------------------------------------------------------------------------
template <int KK, int KS, int PAD, bool OUTF>
__global__ __launch_bounds__(256)
void convgemm(const _Float16* __restrict__ in, int inStride, int Cin,
              const _Float16* __restrict__ w, const float* __restrict__ bias, int OC,
              _Float16* __restrict__ outH, float* __restrict__ outF, int outStride)
{
  __shared__ __align__(16) _Float16 lA[2][64 * LSTR];
  __shared__ __align__(16) _Float16 lB[2][128 * LSTR];

  const int tid  = threadIdx.x;
  const int lane = tid & 31;
  const int wave = tid >> 5;
  const int m0   = (wave & 1) * 32;   // wave's oc quad base within 64
  const int n0   = (wave >> 1) * 32;  // wave's pixel quad base within 128

  const long pix0   = (long)blockIdx.x * 128;  // 128 tokens never cross an image
  const int  b      = (int)(pix0 >> 10);
  const int  pimg   = (int)(pix0 & 1023);
  const int  ocBase = blockIdx.y * 64;

  // Cooperative A load: row tid>>2 (0..63), 8-half part tid&3.
  const int  arow  = tid >> 2;
  const int  apart = (tid & 3) * 8;
  const int  aoc   = ocBase + arow;
  const bool aok   = aoc < OC;
  const long wrowB = (long)(aok ? aoc : (OC - 1)) * (long)(KK * Cin);

  // Cooperative B load: two rows per thread (row tid>>2 and +64).
  const int brow0 = tid >> 2, brow1 = brow0 + 64;
  const int bp0 = pimg + brow0, bp1 = pimg + brow1;
  const int by0 = bp0 >> 5, bx0 = bp0 & 31;
  const int by1 = bp1 >> 5, bx1 = bp1 & 31;

  const int cpk     = Cin >> 5;      // 32-wide chunks per kpos
  const int nchunks = KK * cpk;

  int kp_l = 0, cb_l = 0;            // load cursor
  uint4 av, bv0, bv1;

  auto loadStage = [&](void) {
    // A: clamped row, zero-masked beyond OC
    av = *(const uint4*)(w + wrowB + (long)kp_l * Cin + cb_l + apart);
    if (!aok) av = make_uint4(0u, 0u, 0u, 0u);
    // B: clamped pixel, zero-masked outside the image
    const int ky = kp_l / KS - PAD;        // KS is a compile-time constant
    const int kx = kp_l % KS - PAD;
    const int yy0 = by0 + ky, xx0 = bx0 + kx;
    const int yy1 = by1 + ky, xx1 = bx1 + kx;
    const bool ok0 = (yy0 >= 0) & (yy0 < 32) & (xx0 >= 0) & (xx0 < 32);
    const bool ok1 = (yy1 >= 0) & (yy1 < 32) & (xx1 >= 0) & (xx1 < 32);
    const int q0 = (min(max(yy0, 0), 31) << 5) + min(max(xx0, 0), 31);
    const int q1 = (min(max(yy1, 0), 31) << 5) + min(max(xx1, 0), 31);
    bv0 = *(const uint4*)(in + ((long)b * 1024 + q0) * (long)inStride + cb_l + apart);
    bv1 = *(const uint4*)(in + ((long)b * 1024 + q1) * (long)inStride + cb_l + apart);
    if (!ok0) bv0 = make_uint4(0u, 0u, 0u, 0u);
    if (!ok1) bv1 = make_uint4(0u, 0u, 0u, 0u);
    // advance cursor
    cb_l += 32;
    if (cb_l == Cin) { cb_l = 0; ++kp_l; }
  };

  v8f a00 = {}, a01 = {}, a10 = {}, a11 = {};

  // Preload chunk 0 into buffer 0.
  loadStage();
  *(uint4*)(lA[0] + arow  * LSTR + apart) = av;
  *(uint4*)(lB[0] + brow0 * LSTR + apart) = bv0;
  *(uint4*)(lB[0] + brow1 * LSTR + apart) = bv1;

  int cur = 0;
  for (int ci = 0; ci < nchunks; ++ci) {
    __syncthreads();                       // buf[cur] ready; buf[cur^1] free
    const bool have = (ci + 1) < nchunks;  // block-uniform
    if (have) loadStage();                 // prefetch next chunk (global)
    v16h fa0 = frag_ld(lA[cur], m0,      lane);
    v16h fa1 = frag_ld(lA[cur], m0 + 16, lane);
    v16h fb0 = frag_ld(lB[cur], n0,      lane);
    v16h fb1 = frag_ld(lB[cur], n0 + 16, lane);
    a00 = wmma_f16(fa0, fb0, a00);
    a01 = wmma_f16(fa0, fb1, a01);
    a10 = wmma_f16(fa1, fb0, a10);
    a11 = wmma_f16(fa1, fb1, a11);
    if (have) {
      const int nxt = cur ^ 1;
      *(uint4*)(lA[nxt] + arow  * LSTR + apart) = av;
      *(uint4*)(lB[nxt] + brow0 * LSTR + apart) = bv0;
      *(uint4*)(lB[nxt] + brow1 * LSTR + apart) = bv1;
    }
    cur ^= 1;
  }

  // C/D layout: vgpr r, lanes 0-15 -> M=r, N=lane; lanes 16-31 -> M=r+8, N=lane-16.
  const int mloc = (lane >> 4) << 3;
  const int ncol = lane & 15;
#pragma unroll
  for (int r = 0; r < 8; ++r) {
    const int  ocA = ocBase + m0 + mloc + r;
    const int  ocB = ocA + 16;
    const long tA  = pix0 + n0 + ncol;
    const long tB  = tA + 16;
    const float bA = (ocA < OC) ? bias[ocA] : 0.f;
    const float bB = (ocB < OC) ? bias[ocB] : 0.f;
    if constexpr (OUTF) {
      if (ocA < OC) { outF[tA * outStride + ocA] = a00[r] + bA;
                      outF[tB * outStride + ocA] = a01[r] + bA; }
      if (ocB < OC) { outF[tA * outStride + ocB] = a10[r] + bB;
                      outF[tB * outStride + ocB] = a11[r] + bB; }
    } else {
      if (ocA < OC) { outH[tA * outStride + ocA] = (_Float16)(a00[r] + bA);
                      outH[tB * outStride + ocA] = (_Float16)(a01[r] + bA); }
      if (ocB < OC) { outH[tA * outStride + ocB] = (_Float16)(a10[r] + bB);
                      outH[tB * outStride + ocB] = (_Float16)(a11[r] + bB); }
    }
  }
}

// ---------------------------------------------------------------------------
// Weight reorder + f32->f16: dst[oc][kpos][c] = src[oc][c][kpos]. KK==1 is a
// plain convert (used for x / fc1_w / fc2_w).
// ---------------------------------------------------------------------------
__global__ void reorder_w(const float* __restrict__ src, _Float16* __restrict__ dst,
                          long total, int C, int KK)
{
  long i = (long)blockIdx.x * blockDim.x + threadIdx.x;
  if (i >= total) return;
  long c   = i % C;
  long rem = i / C;
  long kp  = rem % KK;
  long oc  = rem / KK;
  dst[i] = (_Float16)src[(oc * C + c) * (long)KK + kp];
}

// ---------------------------------------------------------------------------
// Deformable bilinear gather (NHWC): one thread handles (token, kpos, 32-ch
// chunk). Corner rows are contiguous 512-ch vectors -> b128 loads.
// ---------------------------------------------------------------------------
__global__ __launch_bounds__(256)
void gather_bilinear(const _Float16* __restrict__ xin, const float* __restrict__ off,
                     _Float16* __restrict__ acc, int KK, int ksize, int pad, long total)
{
  long idx = (long)blockIdx.x * blockDim.x + threadIdx.x;
  if (idx >= total) return;
  const int  cc  = (int)(idx & 15);     // channel chunk (*32)
  long tmp = idx >> 4;
  const int  kp  = (int)(tmp % KK);
  const long t   = tmp / KK;            // token 0..4095
  const int  b   = (int)(t >> 10);
  const int  p   = (int)(t & 1023);
  const int  y   = p >> 5, x = p & 31;
  const int  ky  = kp / ksize, kx = kp % ksize;

  const float dy = off[t * (long)(2 * KK) + kp * 2 + 0];
  const float dx = off[t * (long)(2 * KK) + kp * 2 + 1];
  const float pyf = (float)(y - pad + ky) + dy;
  const float pxf = (float)(x - pad + kx) + dx;
  const float y0f = floorf(pyf), x0f = floorf(pxf);
  const float fy = pyf - y0f, fx = pxf - x0f;
  const int y0 = (int)y0f, x0 = (int)x0f;
  const int y1 = y0 + 1,  x1 = x0 + 1;

  const float v00 = (y0 >= 0 && y0 < 32 && x0 >= 0 && x0 < 32) ? 1.f : 0.f;
  const float v01 = (y0 >= 0 && y0 < 32 && x1 >= 0 && x1 < 32) ? 1.f : 0.f;
  const float v10 = (y1 >= 0 && y1 < 32 && x0 >= 0 && x0 < 32) ? 1.f : 0.f;
  const float v11 = (y1 >= 0 && y1 < 32 && x1 >= 0 && x1 < 32) ? 1.f : 0.f;
  const float w00 = (1.f - fy) * (1.f - fx) * v00;
  const float w01 = (1.f - fy) * fx         * v01;
  const float w10 = fy         * (1.f - fx) * v10;
  const float w11 = fy         * fx         * v11;

  const int cy0 = y0 < 0 ? 0 : (y0 > 31 ? 31 : y0);
  const int cy1 = y1 < 0 ? 0 : (y1 > 31 ? 31 : y1);
  const int cx0 = x0 < 0 ? 0 : (x0 > 31 ? 31 : x0);
  const int cx1 = x1 < 0 ? 0 : (x1 > 31 ? 31 : x1);

  const long base = (long)b * 1024;
  const _Float16* r00 = xin + (base + cy0 * 32 + cx0) * 1024;
  const _Float16* r01 = xin + (base + cy0 * 32 + cx1) * 1024;
  const _Float16* r10 = xin + (base + cy1 * 32 + cx0) * 1024;
  const _Float16* r11 = xin + (base + cy1 * 32 + cx1) * 1024;

  const long dbase = t * (long)(KK * 512) + (long)kp * 512 + cc * 32;
#pragma unroll
  for (int j = 0; j < 4; ++j) {
    const int c = cc * 32 + j * 8;
    half8 h00 = *(const half8*)(r00 + c);
    half8 h01 = *(const half8*)(r01 + c);
    half8 h10 = *(const half8*)(r10 + c);
    half8 h11 = *(const half8*)(r11 + c);
    half8 o;
#pragma unroll
    for (int e = 0; e < 8; ++e) {
      float v = w00 * (float)h00[e] + w01 * (float)h01[e]
              + w10 * (float)h10[e] + w11 * (float)h11[e];
      o[e] = (_Float16)v;
    }
    *(half8*)(acc + dbase + j * 8) = o;
  }
}

// ---------------------------------------------------------------------------
// LayerNorm (over 1024 ch) + exact GELU; one block per token.
// ---------------------------------------------------------------------------
__global__ __launch_bounds__(256)
void ln_gelu(const _Float16* __restrict__ y, const float* __restrict__ g,
             const float* __restrict__ bb, _Float16* __restrict__ out)
{
  __shared__ float s1[256];
  __shared__ float s2[256];
  const long t  = blockIdx.x;
  const int tid = threadIdx.x;
  float vals[4];
  float s = 0.f, sq = 0.f;
#pragma unroll
  for (int j = 0; j < 4; ++j) {
    float v = (float)y[t * 1024 + tid + j * 256];
    vals[j] = v; s += v; sq += v * v;
  }
  s1[tid] = s; s2[tid] = sq;
  __syncthreads();
  for (int st = 128; st > 0; st >>= 1) {
    if (tid < st) { s1[tid] += s1[tid + st]; s2[tid] += s2[tid + st]; }
    __syncthreads();
  }
  const float mu  = s1[0] * (1.f / 1024.f);
  const float var = s2[0] * (1.f / 1024.f) - mu * mu;
  const float inv = rsqrtf(var + 1e-5f);
#pragma unroll
  for (int j = 0; j < 4; ++j) {
    const int c = tid + j * 256;
    float v  = (vals[j] - mu) * inv * g[c] + bb[c];
    float ge = 0.5f * v * (1.f + erff(v * 0.70710678118654752f));
    out[t * 1024 + c] = (_Float16)ge;
  }
}

// ---------------------------------------------------------------------------
extern "C" void kernel_launch(void* const* d_in, const int* in_sizes, int n_in,
                              void* d_out, int out_size, void* d_ws, size_t ws_size,
                              hipStream_t stream)
{
  (void)in_sizes; (void)n_in; (void)out_size; (void)ws_size;
  const float* x       = (const float*)d_in[0];
  // d_in[1]=H, d_in[2]=W (32x32, hardcoded)
  const float* fc1_w   = (const float*)d_in[3];
  const float* fc1_b   = (const float*)d_in[4];
  const float* off3_w  = (const float*)d_in[5];
  const float* off3_b  = (const float*)d_in[6];
  const float* conv3_w = (const float*)d_in[7];
  const float* conv3_b = (const float*)d_in[8];
  const float* off5_w  = (const float*)d_in[9];
  const float* off5_b  = (const float*)d_in[10];
  const float* conv5_w = (const float*)d_in[11];
  const float* conv5_b = (const float*)d_in[12];
  const float* ln_g    = (const float*)d_in[13];
  const float* ln_b    = (const float*)d_in[14];
  const float* fc2_w   = (const float*)d_in[15];
  const float* fc2_b   = (const float*)d_in[16];
  float* out = (float*)d_out;

  // Workspace bump allocator (256B aligned carves, ~157 MB total).
  size_t woff = 0;
  auto carve = [&](size_t bytes) -> void* {
    void* pp = (char*)d_ws + woff;
    woff = (woff + bytes + 255) & ~(size_t)255;
    return pp;
  };
  _Float16* xh    = (_Float16*)carve(4096l * 256 * 2);
  _Float16* w1h   = (_Float16*)carve(1024l * 256 * 2);
  _Float16* w2h   = (_Float16*)carve(256l * 1024 * 2);
  _Float16* wo3   = (_Float16*)carve(18l * 9 * 512 * 2);
  _Float16* wc3   = (_Float16*)carve(512l * 9 * 512 * 2);
  _Float16* wo5   = (_Float16*)carve(50l * 25 * 512 * 2);
  _Float16* wc5   = (_Float16*)carve(512l * 25 * 512 * 2);
  _Float16* h     = (_Float16*)carve(4096l * 1024 * 2);   // NHWC fc1 output
  float*    o3    = (float*)   carve(4096l * 18 * 4);
  float*    o5    = (float*)   carve(4096l * 50 * 4);
  _Float16* accb  = (_Float16*)carve(4096l * 12800 * 2);  // shared by both branches
  _Float16* dbuf  = (_Float16*)carve(4096l * 512 * 2);    // shared by both branches
  _Float16* yfull = (_Float16*)carve(4096l * 1024 * 2);   // concat target
  _Float16* yg    = (_Float16*)carve(4096l * 1024 * 2);   // LN+GELU output

  const dim3 blk(256);
  auto g1 = [](long n) { return dim3((unsigned)((n + 255) / 256)); };

  // f32->f16 conversions / weight reorders (OIHW -> oc,kpos,c)
  reorder_w<<<g1(4096l * 256),    blk, 0, stream>>>(x,       xh,  4096l * 256,    256,  1);
  reorder_w<<<g1(1024l * 256),    blk, 0, stream>>>(fc1_w,   w1h, 1024l * 256,    256,  1);
  reorder_w<<<g1(256l * 1024),    blk, 0, stream>>>(fc2_w,   w2h, 256l * 1024,   1024,  1);
  reorder_w<<<g1(18l * 9 * 512),  blk, 0, stream>>>(off3_w,  wo3, 18l * 9 * 512,  512,  9);
  reorder_w<<<g1(512l * 9 * 512), blk, 0, stream>>>(conv3_w, wc3, 512l * 9 * 512, 512,  9);
  reorder_w<<<g1(50l * 25 * 512), blk, 0, stream>>>(off5_w,  wo5, 50l * 25 * 512, 512, 25);
  reorder_w<<<g1(512l * 25 * 512),blk, 0, stream>>>(conv5_w, wc5, 512l * 25 * 512,512, 25);

  // fc1: h[t,c] = x @ fc1_w^T + b   (x1 = h[:, :512], x2 = h[:, 512:])
  convgemm<1, 1, 0, false><<<dim3(32, 16), blk, 0, stream>>>(
      xh, 256, 256, w1h, fc1_b, 1024, h, nullptr, 1024);
  // ---- branch k=3 ----
  convgemm<9, 3, 1, true><<<dim3(32, 1), blk, 0, stream>>>(
      h, 1024, 512, wo3, off3_b, 18, nullptr, o3, 18);
  gather_bilinear<<<g1(4096l * 9 * 16), blk, 0, stream>>>(h, o3, accb, 9, 3, 1,
                                                          4096l * 9 * 16);
  convgemm<1, 1, 0, false><<<dim3(32, 8), blk, 0, stream>>>(
      accb, 4608, 4608, wc3, conv3_b, 512, dbuf, nullptr, 512);
  convgemm<9, 3, 1, false><<<dim3(32, 8), blk, 0, stream>>>(
      dbuf, 512, 512, wc3, conv3_b, 512, yfull, nullptr, 1024);
  // ---- branch k=5 ----
  convgemm<25, 5, 2, true><<<dim3(32, 1), blk, 0, stream>>>(
      h + 512, 1024, 512, wo5, off5_b, 50, nullptr, o5, 50);
  gather_bilinear<<<g1(4096l * 25 * 16), blk, 0, stream>>>(h + 512, o5, accb, 25, 5, 2,
                                                           4096l * 25 * 16);
  convgemm<1, 1, 0, false><<<dim3(32, 8), blk, 0, stream>>>(
      accb, 12800, 12800, wc5, conv5_b, 512, dbuf, nullptr, 512);
  convgemm<25, 5, 2, false><<<dim3(32, 8), blk, 0, stream>>>(
      dbuf, 512, 512, wc5, conv5_b, 512, yfull + 512, nullptr, 1024);
  // LN + GELU
  ln_gelu<<<dim3(4096), blk, 0, stream>>>(yfull, ln_g, ln_b, yg);
  // fc2 -> final f32 output
  convgemm<1, 1, 0, true><<<dim3(32, 4), blk, 0, stream>>>(
      yg, 1024, 1024, w2h, fc2_b, 256, nullptr, out, 256);
}
